// JIRANO_87600152969629
// MI455X (gfx1250) — matrix operand order repaired and checked by reading, not compiled, not run on hardware
//
#include <hip/hip_runtime.h>
#include <hip/hip_bf16.h>

typedef __attribute__((ext_vector_type(16))) _Float16 v16h;
typedef __attribute__((ext_vector_type(8)))  float    v8f;

// Problem constants
#define BATCH 16
#define CDIM  384
#define HW    3136              // 56*56
#define NROWS (BATCH*HW)        // 50176
#define KCODE 1024
#define ROWT  16                // rows per tile
#define NROWT (NROWS/ROWT)      // 3136 row tiles (196 per batch image)
#define KT1   (CDIM/32)         // 12  k-steps GEMM1
#define NT1   (KCODE/16)        // 64  n-tiles GEMM1
#define KT2   (KCODE/32)        // 32  k-steps GEMM2
#define NT2   (CDIM/16)         // 24  n-tiles GEMM2

#define B1_STAGE_BYTES (NT1 * 512 * 2)   // 65536 B: one full k-step of B1 frags

// ---------------- async copy to LDS (CDNA5 path, guarded) ----------------
#if defined(__has_builtin)
#if __has_builtin(__builtin_amdgcn_global_load_async_to_lds_b128)
#define HAVE_ASYNC_LDS 1
#endif
#if __has_builtin(__builtin_amdgcn_s_wait_asynccnt)
#define HAVE_WAIT_ASYNC 1
#endif
#endif

typedef int b128v __attribute__((vector_size(16)));   // matches builtin param pointee

__device__ __forceinline__ void copy16_g2l(const void* g, void* l) {
#if defined(HAVE_ASYNC_LDS)
    __builtin_amdgcn_global_load_async_to_lds_b128(
        (__attribute__((address_space(1))) b128v*)(void*)g,
        (__attribute__((address_space(3))) b128v*)l, 0, 0);
#else
    *(int4*)l = *(const int4*)g;       // global_load_b128 + ds_store_b128
#endif
}

__device__ __forceinline__ void wait_async0() {
#if defined(HAVE_ASYNC_LDS)
#if defined(HAVE_WAIT_ASYNC)
    __builtin_amdgcn_s_wait_asynccnt(0);
#else
    asm volatile("s_wait_asynccnt 0" ::: "memory");
#endif
#endif
}

// ---- WMMA fragment element maps (wave32, 16x16x32 f16) ----
// A (16x32): lane<16 holds row M=lane, K in {0..7,16..23}; lane>=16 row lane-16, K in {8..15,24..31}
__device__ __forceinline__ int aKmap(int lane, int j) {
    return ((j >> 3) << 4) | ((lane >> 4) << 3) | (j & 7);
}
// B (32x16): lane holds col N=lane&15; K in {0..7,16..23} (lane<16) or {8..15,24..31} (lane>=16)
__device__ __forceinline__ int bKmap(int lane, int j) {
    return ((j >> 3) << 4) | ((lane >> 4) << 3) | (j & 7);
}

union PK2 { _Float16 h[2]; unsigned u; };

// =====================================================================
// Kernel: codebook prep — B fragments for both GEMMs + ||w||^2
// =====================================================================
__global__ __launch_bounds__(256) void k_prep_w(const float* __restrict__ vq,
                                                _Float16* __restrict__ b1,
                                                _Float16* __restrict__ b2,
                                                float* __restrict__ wnorm) {
    int bid = blockIdx.x, tid = threadIdx.x;
    if (bid < KT1 * NT1) {                       // B1: B[kc][n] = W[n][kc]
        int e = tid * 2, lane = e >> 4, j = e & 15;   // j even
        int nt = bid % NT1;
        int kt = bid / NT1;
        int code = nt * 16 + (lane & 15);
        int c0 = kt * 32 + bKmap(lane, j);       // bK(lane,j+1)=bK+1
        PK2 pk;
        pk.h[0] = (_Float16)vq[(size_t)code * CDIM + c0];
        pk.h[1] = (_Float16)vq[(size_t)code * CDIM + c0 + 1];
        *(unsigned*)((char*)b1 + ((size_t)bid * 512 + e) * 2) = pk.u;
    } else if (bid < KT1 * NT1 + KT2 * NT2) {    // B2: B[kcode][c] = W[kcode][c]
        int t = bid - KT1 * NT1;
        int e = tid * 2, lane = e >> 4, j = e & 15;
        int nt = t % NT2;
        int kt = t / NT2;
        int code0 = kt * 32 + bKmap(lane, j);
        int c = nt * 16 + (lane & 15);
        PK2 pk;
        pk.h[0] = (_Float16)vq[(size_t)code0 * CDIM + c];
        pk.h[1] = (_Float16)vq[(size_t)(code0 + 1) * CDIM + c];
        *(unsigned*)((char*)b2 + ((size_t)t * 512 + e) * 2) = pk.u;
    } else {                                     // wnorm
        int code = (bid - (KT1 * NT1 + KT2 * NT2)) * 256 + tid;
        const float* row = vq + (size_t)code * CDIM;
        float s = 0.f;
        for (int c = 0; c < CDIM; ++c) { float v = row[c]; s += v * v; }
        wnorm[code] = s;
    }
}

// =====================================================================
// Kernel: x prep — LDS transpose, feat_bhwc out, A-frags (f16), ||x||^2
// =====================================================================
__global__ __launch_bounds__(256) void k_prep_x(const float* __restrict__ feat,
                                                float* __restrict__ out_feat,
                                                _Float16* __restrict__ aFrag,
                                                float* __restrict__ xnorm) {
    __shared__ float ldsX[ROWT * 388];
    __shared__ float snorm[ROWT];
    int tid = threadIdx.x;
    int rowTile = blockIdx.x;
    int bIdx = rowTile / 196;
    int hw0 = (rowTile % 196) * 16;
    int tx = tid & 15, ty = tid >> 4;

    const float* fbase = feat + (size_t)bIdx * CDIM * HW + hw0;
    for (int c0 = 0; c0 < CDIM; c0 += 16) {
        int c = c0 + ty;
        ldsX[tx * 388 + c] = fbase[(size_t)c * HW + tx];
    }
    if (tid < ROWT) snorm[tid] = 0.f;
    __syncthreads();

    float* ob = out_feat + (size_t)rowTile * ROWT * CDIM;
    for (int i = tid; i < ROWT * CDIM; i += 256) {
        int r = i / CDIM, c = i - r * CDIM;
        ob[i] = ldsX[r * 388 + c];
    }

    {
        int r = tid & 15, cs = tid >> 4;
        float s = 0.f;
        for (int c = cs; c < CDIM; c += 16) { float v = ldsX[r * 388 + c]; s += v * v; }
        atomicAdd(&snorm[r], s);    // ds_add_f32
    }

    _Float16* af = aFrag + (size_t)rowTile * KT1 * 512;
    int e = tid * 2, lane = e >> 4, j = e & 15;   // j even
    int M = lane & 15;
    for (int kt = 0; kt < KT1; ++kt) {
        int K0 = kt * 32 + aKmap(lane, j);
        PK2 pk;
        pk.h[0] = (_Float16)ldsX[M * 388 + K0];
        pk.h[1] = (_Float16)ldsX[M * 388 + K0 + 1];
        *(unsigned*)((char*)af + ((size_t)kt * 512 + e) * 2) = pk.u;
    }

    __syncthreads();
    if (tid < ROWT) xnorm[rowTile * ROWT + tid] = snorm[tid];
}

// =====================================================================
// Kernel: GEMM1  distance = ||x||^2 + ||w||^2 - 2 * x @ Wᵀ
// 8 waves; B1 frags async-staged into a 2x64KB LDS double buffer.
// Per k-step: burst-load all 8 B tiles into distinct registers, then
// drain with 8 WMMAs (descending non-zero ds waits); async stage of the
// next k-step's slab runs concurrently.
// =====================================================================
__global__ __launch_bounds__(256) void k_dist_gemm(const _Float16* __restrict__ aFrag,
                                                   const _Float16* __restrict__ b1,
                                                   const float* __restrict__ xnorm,
                                                   const float* __restrict__ wnorm,
                                                   float* __restrict__ dist) {
    extern __shared__ char smem[];               // 2 * 65536
    int tid = threadIdx.x, lane = tid & 31, wv = tid >> 5;
    int rowTile = blockIdx.x;
    const _Float16* aBase = aFrag + (size_t)rowTile * KT1 * 512 + lane * 16;

    // prologue: stage k-step 0
    {
        const char* g = (const char*)b1;
#pragma unroll
        for (int i = 0; i < 16; ++i)
            copy16_g2l(g + i * 4096 + tid * 16, smem + i * 4096 + tid * 16);
    }
    wait_async0();
    __syncthreads();

    const v8f vzero = {0.f, 0.f, 0.f, 0.f, 0.f, 0.f, 0.f, 0.f};
    v8f acc[8];
#pragma unroll
    for (int i = 0; i < 8; ++i) acc[i] = vzero;

    v16h aCur = *(const v16h*)aBase;             // kt=0 A frag

    for (int kt = 0; kt < KT1; ++kt) {
        // stage next k-step while computing this one
        if (kt + 1 < KT1) {
            const char* g = (const char*)(b1 + (size_t)(kt + 1) * NT1 * 512);
            char* l = smem + ((kt + 1) & 1) * B1_STAGE_BYTES;
#pragma unroll
            for (int i = 0; i < 16; ++i)
                copy16_g2l(g + i * 4096 + tid * 16, l + i * 4096 + tid * 16);
        }
        v16h aNext;
        if (kt + 1 < KT1) aNext = *(const v16h*)(aBase + (kt + 1) * 512);

        // burst-load 8 B tiles into distinct regs, then 8 WMMAs
        const char* l = smem + (kt & 1) * B1_STAGE_BYTES + (size_t)(wv * 8) * 1024 + lane * 32;
        v16h breg[8];
#pragma unroll
        for (int nt = 0; nt < 8; ++nt)
            breg[nt] = *(const v16h*)(l + nt * 1024);
#pragma unroll
        for (int nt = 0; nt < 8; ++nt)
            acc[nt] = __builtin_amdgcn_wmma_f32_16x16x32_f16(
                false, aCur, false, breg[nt], (short)0, acc[nt], false, false);

        if (kt + 1 < KT1) {
            aCur = aNext;
            wait_async0();
            __syncthreads();                     // rotate double buffer
        }
    }

    // Epilogue: fuse norms, write distance (64B-contiguous per half-wave)
    int rHalf = (lane < 16) ? 0 : 8;
    int rbase = rowTile * ROWT + rHalf;
    float xn[8];
#pragma unroll
    for (int r = 0; r < 8; ++r) xn[r] = xnorm[rbase + r];
    int cl = lane & 15;
#pragma unroll
    for (int nt = 0; nt < 8; ++nt) {
        int col = wv * 128 + nt * 16 + cl;
        float wn = wnorm[col];
#pragma unroll
        for (int r = 0; r < 8; ++r) {
            dist[(size_t)(rbase + r) * KCODE + col] = fmaf(-2.f, acc[nt][r], xn[r] + wn);
        }
    }
}

// =====================================================================
// Kernel: softmax + assignment + GEMM2 (q = p @ W) + q_feat
// =====================================================================
#define SD_STRIDE 1041
#define SMEM_PFRAG 66624
#define SMEM_QTILE (66624 + 32768)
#define SMEM_TOTAL (66624 + 32768 + 8704)

__global__ __launch_bounds__(256) void k_softmax_gemm2(const float* __restrict__ dist,
                                                       const _Float16* __restrict__ b2,
                                                       float* __restrict__ out_assign,
                                                       float* __restrict__ out_qfeat) {
    extern __shared__ char smem[];
    float*    sdist = (float*)smem;
    _Float16* pfrag = (_Float16*)(smem + SMEM_PFRAG);
    float*    qtile = (float*)(smem + SMEM_QTILE);
    __shared__ float rmin[ROWT], rinv[ROWT];

    int tid = threadIdx.x, lane = tid & 31, wv = tid >> 5;
    int rowTile = blockIdx.x;
    int bIdx = rowTile / 196;
    int hw0 = (rowTile % 196) * 16;

    const float* dsrc = dist + (size_t)rowTile * ROWT * KCODE;
    for (int i = tid; i < ROWT * KCODE; i += 256) {
        int r = i >> 10, k = i & 1023;
        sdist[r * SD_STRIDE + k] = dsrc[i];
    }
    __syncthreads();

    // Row stats: each wave reduces 2 rows via shuffles
    for (int rr = 0; rr < 2; ++rr) {
        int r = wv * 2 + rr;
        const float* dr = sdist + r * SD_STRIDE;
        float mn = 3.4e38f;
        for (int i = lane; i < KCODE; i += 32) mn = fminf(mn, dr[i]);
#pragma unroll
        for (int o = 16; o > 0; o >>= 1) mn = fminf(mn, __shfl_xor(mn, o, 32));
        float s = 0.f;
        for (int i = lane; i < KCODE; i += 32) s += __expf(mn - dr[i]);
#pragma unroll
        for (int o = 16; o > 0; o >>= 1) s += __shfl_xor(s, o, 32);
        if (lane == 0) { rmin[r] = mn; rinv[r] = 1.f / s; }
    }
    __syncthreads();

    // p = softmax(-d); write assignment (b,K,h,w): 16 rows = one 64B hw line per k
    size_t abase = (size_t)bIdx * KCODE * HW + hw0;
    for (int i = tid; i < ROWT * KCODE; i += 256) {
        int r = i & 15, k = i >> 4;
        float p = __expf(rmin[r] - sdist[r * SD_STRIDE + k]) * rinv[r];
        sdist[r * SD_STRIDE + k] = p;
        out_assign[abase + (size_t)k * HW + r] = p;
    }
    __syncthreads();

    // Convert p -> f16 A fragments in LDS
    for (int e2 = tid; e2 < KT2 * 256; e2 += 256) {
        int e = e2 * 2;
        int kt = e >> 9, rem = e & 511, ln = rem >> 4, j = rem & 15;  // j even
        int M = ln & 15;
        int K0 = kt * 32 + aKmap(ln, j);
        PK2 pk;
        pk.h[0] = (_Float16)sdist[M * SD_STRIDE + K0];
        pk.h[1] = (_Float16)sdist[M * SD_STRIDE + K0 + 1];
        *(unsigned*)((char*)pfrag + e * 2) = pk.u;
    }
    __syncthreads();

    // GEMM2: q(16x384) = p(16x1024) @ W(1024x384); wave owns 3 n-tiles.
    // B2 regs and pfrag A both prefetched one k-step ahead.
    const v8f vzero = {0.f, 0.f, 0.f, 0.f, 0.f, 0.f, 0.f, 0.f};
    v8f acc[3];
#pragma unroll
    for (int i = 0; i < 3; ++i) acc[i] = vzero;

    const _Float16* bWave = b2 + (size_t)(wv * 3) * 512 + lane * 16;
    v16h bcur[3], bnxt[3];
#pragma unroll
    for (int i = 0; i < 3; ++i) bcur[i] = *(const v16h*)(bWave + (size_t)i * 512);
    v16h aCur = *(const v16h*)(pfrag + lane * 16);            // kt=0 ds_load

    for (int kt = 0; kt < KT2; ++kt) {
        v16h aNext;
        if (kt + 1 < KT2) {
            aNext = *(const v16h*)(pfrag + (kt + 1) * 512 + lane * 16);
#pragma unroll
            for (int i = 0; i < 3; ++i)
                bnxt[i] = *(const v16h*)(bWave + ((size_t)(kt + 1) * NT2 + i) * 512);
        }
#pragma unroll
        for (int i = 0; i < 3; ++i) {
            acc[i] = __builtin_amdgcn_wmma_f32_16x16x32_f16(
                false, aCur, false, bcur[i], (short)0, acc[i], false, false);
        }
        if (kt + 1 < KT2) {
            aCur = aNext;
#pragma unroll
            for (int i = 0; i < 3; ++i) bcur[i] = bnxt[i];
        }
    }

    // Epilogue: per-wave LDS transpose staging, hw-coalesced q_feat stores
    float* qt = qtile + wv * 272;
    int rHalf = (lane < 16) ? 0 : 8;
    int cl = lane & 15;
#pragma unroll
    for (int i = 0; i < 3; ++i) {
        int c0 = (wv * 3 + i) * 16;
#pragma unroll
        for (int r = 0; r < 8; ++r) qt[cl * 17 + rHalf + r] = acc[i][r];
        asm volatile("s_wait_dscnt 0" ::: "memory");
        size_t qb = ((size_t)bIdx * CDIM + c0) * HW + hw0;
#pragma unroll
        for (int cc = 0; cc < 8; ++cc) {
            int cL = cc * 2 + (lane >> 4);
            int r = lane & 15;
            out_qfeat[qb + (size_t)cL * HW + r] = qt[cL * 17 + r];
        }
        asm volatile("s_wait_dscnt 0" ::: "memory");
    }
}

// =====================================================================
extern "C" void kernel_launch(void* const* d_in, const int* in_sizes, int n_in,
                              void* d_out, int out_size, void* d_ws, size_t ws_size,
                              hipStream_t stream) {
    const float* feat = (const float*)d_in[0];   // (16,384,56,56) f32
    const float* vq   = (const float*)d_in[1];   // (1024,384) f32

    float* out = (float*)d_out;
    const size_t S_feat   = (size_t)BATCH * HW * CDIM;   // 19,267,584
    const size_t S_assign = (size_t)BATCH * KCODE * HW;  // 51,380,224
    float* out_feat   = out;
    float* out_qfeat  = out + S_feat;
    float* out_assign = out + 2 * S_feat;
    float* out_dist   = out + 2 * S_feat + S_assign;

    char* ws = (char*)d_ws;
    const size_t SZ_AFRAG = (size_t)NROWT * KT1 * 512 * 2;   // 38,535,168
    const size_t SZ_B1    = (size_t)KT1 * NT1 * 512 * 2;     //    786,432
    const size_t SZ_B2    = (size_t)KT2 * NT2 * 512 * 2;     //    786,432
    const size_t SZ_XN    = (size_t)NROWS * 4;               //    200,704
    _Float16* aFrag = (_Float16*)ws;
    _Float16* b1    = (_Float16*)(ws + SZ_AFRAG);
    _Float16* b2    = (_Float16*)(ws + SZ_AFRAG + SZ_B1);
    float*    xnorm = (float*)   (ws + SZ_AFRAG + SZ_B1 + SZ_B2);
    float*    wnorm = (float*)   (ws + SZ_AFRAG + SZ_B1 + SZ_B2 + SZ_XN);

    hipLaunchKernelGGL(k_prep_w, dim3(KT1 * NT1 + KT2 * NT2 + KCODE / 256), dim3(256),
                       0, stream, vq, b1, b2, wnorm);
    hipLaunchKernelGGL(k_prep_x, dim3(NROWT), dim3(256), 0, stream,
                       feat, out_feat, aFrag, xnorm);
    hipLaunchKernelGGL(k_dist_gemm, dim3(NROWT), dim3(256), 2 * B1_STAGE_BYTES, stream,
                       aFrag, b1, xnorm, wnorm, out_dist);
    hipLaunchKernelGGL(k_softmax_gemm2, dim3(NROWT), dim3(256), SMEM_TOTAL, stream,
                       out_dist, b2, out_assign, out_qfeat);
}
